// MultiHeadSelfAttention_55834574848209
// MI455X (gfx1250) — matrix operand order repaired and checked by reading, not compiled, not run on hardware
//
#include <hip/hip_runtime.h>
#include <hip/hip_bf16.h>

typedef __bf16 bf16;
typedef __attribute__((ext_vector_type(16))) __bf16 v16bf;
typedef __attribute__((ext_vector_type(8)))  __bf16 v8bf;
typedef __attribute__((ext_vector_type(8)))  float  v8f;
typedef __attribute__((ext_vector_type(4)))  int    v4i;

union Frag16 { v16bf v; v8bf h[2]; };

__device__ __forceinline__ v8f wmma_bf16(const Frag16& a, const Frag16& b, v8f c) {
    return __builtin_amdgcn_wmma_f32_16x16x32_bf16(
        /*neg_a=*/false, a.v, /*neg_b=*/false, b.v,
        /*c_mod=*/(short)0, c, /*reuse_a=*/false, /*reuse_b=*/false);
}

// ---------------------------------------------------------------------------
// Async global->LDS staging (CDNA5 GLOBAL_LOAD_ASYNC_TO_LDS_B128, ASYNCcnt).
// Builtin expects (as(1) v4i*, as(3) v4i*, imm offset, imm cpol).
// Guarded by __has_builtin; falls back to load+ds_store staging otherwise.
// ---------------------------------------------------------------------------
#if defined(__has_builtin)
#if __has_builtin(__builtin_amdgcn_global_load_async_to_lds_b128)
#define HAVE_ASYNC_LDS 1
#endif
#endif

#ifdef HAVE_ASYNC_LDS
__device__ __forceinline__ void async_copy16(const bf16* g, bf16* l) {
    __builtin_amdgcn_global_load_async_to_lds_b128(
        (__attribute__((address_space(1))) v4i*)
            (__attribute__((address_space(1))) void*)(g),
        (__attribute__((address_space(3))) v4i*)
            (__attribute__((address_space(3))) void*)(l),
        /*offset=*/0, /*cpol=*/0);
}
__device__ __forceinline__ void async_wait0() {
#if __has_builtin(__builtin_amdgcn_s_wait_asynccnt)
    __builtin_amdgcn_s_wait_asynccnt(0);
#else
    asm volatile("s_wait_asynccnt 0" ::: "memory");
#endif
}
#else
__device__ __forceinline__ void async_copy16(const bf16* g, bf16* l) {
    *reinterpret_cast<v8bf*>(l) = *reinterpret_cast<const v8bf*>(g);
}
__device__ __forceinline__ void async_wait0() {}
#endif

// ---------------------------------------------------------------------------
// DPP16 butterfly reductions across the 16-lane half-rows (lanes 0-15 and
// 16-31 reduce independently — matching the WMMA C-tile row layout).
// quad_perm(1,0,3,2)=xor1, quad_perm(2,3,0,1)=xor2, row_half_mirror=xor7,
// row_mirror=xor15. VALU-only: no LDS traffic, no dscnt waits.
// ---------------------------------------------------------------------------
template<int CTRL>
__device__ __forceinline__ float dpp_mov_f32(float x) {
    return __builtin_bit_cast(float, __builtin_amdgcn_update_dpp(
        0, __builtin_bit_cast(int, x), CTRL, 0xF, 0xF, true));
}
__device__ __forceinline__ float red_max16(float x) {
    x = fmaxf(x, dpp_mov_f32<0x0B1>(x));   // quad_perm(1,0,3,2)
    x = fmaxf(x, dpp_mov_f32<0x04E>(x));   // quad_perm(2,3,0,1)
    x = fmaxf(x, dpp_mov_f32<0x141>(x));   // row_half_mirror
    x = fmaxf(x, dpp_mov_f32<0x140>(x));   // row_mirror
    return x;
}
__device__ __forceinline__ float red_sum16(float x) {
    x += dpp_mov_f32<0x0B1>(x);
    x += dpp_mov_f32<0x04E>(x);
    x += dpp_mov_f32<0x141>(x);
    x += dpp_mov_f32<0x140>(x);
    return x;
}

// ---------------------------------------------------------------------------
// Kernel 0: f32 -> bf16 conversion, 8 elements per thread (16B packed stores)
// ---------------------------------------------------------------------------
__global__ __launch_bounds__(256) void cvt_bf16_kernel(
    const float* __restrict__ src, bf16* __restrict__ dst)
{
    size_t i = ((size_t)blockIdx.x * 256 + threadIdx.x) * 8;
    float4 f0 = *reinterpret_cast<const float4*>(src + i);
    float4 f1 = *reinterpret_cast<const float4*>(src + i + 4);
    v8bf o;
    o[0] = (bf16)f0.x; o[1] = (bf16)f0.y; o[2] = (bf16)f0.z; o[3] = (bf16)f0.w;
    o[4] = (bf16)f1.x; o[5] = (bf16)f1.y; o[6] = (bf16)f1.z; o[7] = (bf16)f1.w;
    *reinterpret_cast<v8bf*>(dst + i) = o;
}

// ---------------------------------------------------------------------------
// Generic WMMA GEMM: C[M,N] = A[M,K] * Bw[N,K]^T   (A,Bw bf16 row-major)
// Block tile 128x128, BK=32, 256 threads = 8 waves (4 in M x 2 in N),
// each wave computes a 32x64 region = 2x4 WMMA tiles.
// Tile staging goes straight to LDS via async copies (no VGPR round trip).
// ---------------------------------------------------------------------------
template<bool OUT_BF16>
__global__ __launch_bounds__(256) void gemm_wmma_kernel(
    const bf16* __restrict__ A, const bf16* __restrict__ Bw,
    void* __restrict__ Cout, int M, int N, int K)
{
    __shared__ bf16 As[128][40];   // 32 cols + 8 pad (16B-aligned rows, conflict-free)
    __shared__ bf16 Bs[128][40];

    const int tid  = threadIdx.x;
    const int lane = tid & 31;
    const int w    = tid >> 5;
    const int wm   = w & 3;        // 0..3  -> 32-row strip
    const int wn   = w >> 2;       // 0..1  -> 64-col strip
    const int ln   = lane & 15;
    const int half = lane >> 4;
    const int m0   = blockIdx.y * 128;
    const int n0   = blockIdx.x * 128;

    v8f acc[2][4] = {};

    for (int k0 = 0; k0 < K; k0 += 32) {
        __syncthreads();
        // Stage 128x32 bf16 tiles of A and B: 512 16B-chunks each, 2+2 per thread.
#pragma unroll
        for (int s = 0; s < 2; ++s) {
            int c   = tid + s * 256;
            int row = c >> 2;
            int col = (c & 3) * 8;
            async_copy16(&A [(size_t)(m0 + row) * K + k0 + col], &As[row][col]);
            async_copy16(&Bw[(size_t)(n0 + row) * K + k0 + col], &Bs[row][col]);
#ifndef HAVE_ASYNC_LDS
            if (k0 + 32 < K) {   // prefetch next K tile -> global_prefetch
                __builtin_prefetch(&A [(size_t)(m0 + row) * K + k0 + 32 + col], 0, 1);
                __builtin_prefetch(&Bw[(size_t)(n0 + row) * K + k0 + 32 + col], 0, 1);
            }
#endif
        }
        async_wait0();
        __syncthreads();

        // A fragments (16x32): lane(ln,half) holds K = {8h..8h+7, 16+8h..16+8h+7}
        Frag16 af[2];
#pragma unroll
        for (int i = 0; i < 2; ++i) {
            int r = wm * 32 + i * 16 + ln;
            af[i].h[0] = *reinterpret_cast<const v8bf*>(&As[r][half * 8]);
            af[i].h[1] = *reinterpret_cast<const v8bf*>(&As[r][16 + half * 8]);
        }
        // B fragments (32x16): lane(ln,half) holds K = {8h..8h+7, 16+8h..16+8h+7} of column ln
#pragma unroll
        for (int jn = 0; jn < 4; ++jn) {
            Frag16 bfg;
            int r = wn * 64 + jn * 16 + ln;
            bfg.h[0] = *reinterpret_cast<const v8bf*>(&Bs[r][8 * half]);
            bfg.h[1] = *reinterpret_cast<const v8bf*>(&Bs[r][16 + 8 * half]);
#pragma unroll
            for (int i = 0; i < 2; ++i)
                acc[i][jn] = wmma_bf16(af[i], bfg, acc[i][jn]);
        }
    }

    // Epilogue: C tile layout — VGPR j holds row m = 8*half + j, col = ln
#pragma unroll
    for (int i = 0; i < 2; ++i)
#pragma unroll
        for (int jn = 0; jn < 4; ++jn)
#pragma unroll
            for (int j = 0; j < 8; ++j) {
                int gm = m0 + wm * 32 + i * 16 + half * 8 + j;
                int gn = n0 + wn * 64 + jn * 16 + ln;
                float v = acc[i][jn][j];
                if (OUT_BF16)
                    ((bf16*)Cout)[(size_t)gm * N + gn] = (bf16)v;
                else
                    ((float*)Cout)[(size_t)gm * N + gn] = v;
            }
}

// ---------------------------------------------------------------------------
// Flash attention: grid = (S/128, B*H). Block = 8 waves, each wave owns 16
// queries. Online softmax over 32-key tiles; all matmuls via WMMA bf16,
// row reductions via DPP16, K-tile staged with async global->LDS copies.
// qkv layout: (B*S, 3072) bf16 with f = sel*1024 + h*64 + d.
// aout layout: (B*S, 1024) bf16 with a = h*64 + d.
// ---------------------------------------------------------------------------
__global__ __launch_bounds__(256) void attn_kernel(
    const bf16* __restrict__ qkv, bf16* __restrict__ aout)
{
    constexpr int S = 2048, DH = 64, TA = 3072;
    __shared__ bf16 Ks[32][72];       // 32 keys x 64 dh (+8 pad)
    __shared__ bf16 Vt[64][40];       // transposed: d-major (32 keys +8 pad)
    __shared__ bf16 Ps[8][16][40];    // per-wave P staging, row-major 16x32 (+8 pad)

    const int tid  = threadIdx.x;
    const int lane = tid & 31;
    const int w    = tid >> 5;
    const int ln   = lane & 15;
    const int half = lane >> 4;
    const int bh   = blockIdx.y;
    const int b    = bh >> 4;
    const int h    = bh & 15;
    const int q0   = blockIdx.x * 128;

    const size_t base = (size_t)b * S * TA;
    const bf16* Qp = qkv + base + 0    + h * DH;
    const bf16* Kp = qkv + base + 1024 + h * DH;
    const bf16* Vp = qkv + base + 2048 + h * DH;

    // Q fragments: 16 rows x 64 dh as two 16x32 A-matrices (held in registers)
    Frag16 qf[2];
    {
        size_t r = (size_t)(q0 + w * 16 + ln) * TA;
#pragma unroll
        for (int c = 0; c < 2; ++c) {
            qf[c].h[0] = *reinterpret_cast<const v8bf*>(&Qp[r + c * 32 + half * 8]);
            qf[c].h[1] = *reinterpret_cast<const v8bf*>(&Qp[r + c * 32 + 16 + half * 8]);
        }
    }

    v8f   o[4] = {};
    float run_max[8], run_sum[8];
#pragma unroll
    for (int j = 0; j < 8; ++j) { run_max[j] = -1e30f; run_sum[j] = 0.0f; }

    const float c_exp = 0.125f * 1.44269504088896f;   // dh^-0.5 * log2(e)

    for (int kt = 0; kt < S; kt += 32) {
        __syncthreads();
        {   // stage K tile (async, row-major) and V tile (manual transpose)
            int row = tid >> 3;
            int col = (tid & 7) * 8;
            async_copy16(&Kp[(size_t)(kt + row) * TA + col], &Ks[row][col]);
            v8bf vv = *reinterpret_cast<const v8bf*>(&Vp[(size_t)(kt + row) * TA + col]);
#pragma unroll
            for (int i = 0; i < 8; ++i) Vt[col + i][row] = vv[i];
        }
        async_wait0();
        __syncthreads();

        // scores: two 16x16 tiles (keys kt..kt+15, kt+16..kt+31), K-dim = 64
        v8f s[2];
#pragma unroll
        for (int t = 0; t < 2; ++t) {
            v8f z = {};
            int kr = t * 16 + ln;
            Frag16 kf;
#pragma unroll
            for (int c = 0; c < 2; ++c) {
                kf.h[0] = *reinterpret_cast<const v8bf*>(&Ks[kr][c * 32 + half * 8]);
                kf.h[1] = *reinterpret_cast<const v8bf*>(&Ks[kr][c * 32 + 16 + half * 8]);
                z = wmma_bf16(qf[c], kf, z);
            }
            s[t] = z;
        }

        // online softmax; row m = 8*half + j lives across the 16 lanes of a half
        float corr[8];
#pragma unroll
        for (int j = 0; j < 8; ++j) {
            float mx = red_max16(fmaxf(s[0][j], s[1][j]));
            float m_new = fmaxf(run_max[j], mx);
            float cf = __builtin_amdgcn_exp2f((run_max[j] - m_new) * c_exp);
            float p0 = __builtin_amdgcn_exp2f((s[0][j] - m_new) * c_exp);
            float p1 = __builtin_amdgcn_exp2f((s[1][j] - m_new) * c_exp);
            s[0][j] = p0; s[1][j] = p1;
            float ps = red_sum16(p0 + p1);
            run_max[j] = m_new;
            run_sum[j] = run_sum[j] * cf + ps;
            corr[j] = cf;
        }

        // P: C-layout f32 -> row-major bf16 in per-wave LDS, reload as A frag
#pragma unroll
        for (int j = 0; j < 8; ++j) {
            int m = half * 8 + j;
            Ps[w][m][ln]      = (bf16)s[0][j];
            Ps[w][m][16 + ln] = (bf16)s[1][j];
        }
        // rescale accumulators while the P stores land (DS is in-order per wave)
#pragma unroll
        for (int dg = 0; dg < 4; ++dg)
#pragma unroll
            for (int j = 0; j < 8; ++j) o[dg][j] *= corr[j];

        Frag16 pf;
        pf.h[0] = *reinterpret_cast<const v8bf*>(&Ps[w][ln][half * 8]);
        pf.h[1] = *reinterpret_cast<const v8bf*>(&Ps[w][ln][16 + half * 8]);

        // O += P(16x32) x V(32x64); V B-frags contiguous thanks to the transpose
#pragma unroll
        for (int dg = 0; dg < 4; ++dg) {
            Frag16 vf;
            int dr = dg * 16 + ln;
            vf.h[0] = *reinterpret_cast<const v8bf*>(&Vt[dr][half * 8]);
            vf.h[1] = *reinterpret_cast<const v8bf*>(&Vt[dr][16 + half * 8]);
            o[dg] = wmma_bf16(pf, vf, o[dg]);
        }
    }

    // normalize and write bf16 attention output
    float inv[8];
#pragma unroll
    for (int j = 0; j < 8; ++j) inv[j] = 1.0f / run_sum[j];
    bf16* op = aout + ((size_t)b * S + q0 + w * 16) * 1024 + h * DH;
#pragma unroll
    for (int dg = 0; dg < 4; ++dg)
#pragma unroll
        for (int j = 0; j < 8; ++j) {
            int m = half * 8 + j;
            op[(size_t)m * 1024 + dg * 16 + ln] = (bf16)(o[dg][j] * inv[j]);
        }
}

// ---------------------------------------------------------------------------
// Launch: convert -> QKV GEMM -> attention -> output GEMM
// ---------------------------------------------------------------------------
extern "C" void kernel_launch(void* const* d_in, const int* in_sizes, int n_in,
                              void* d_out, int out_size, void* d_ws, size_t ws_size,
                              hipStream_t stream) {
    const float* x     = (const float*)d_in[0];   // (2,2048,1024)
    const float* w_qkv = (const float*)d_in[1];   // (3072,1024)
    const float* w_out = (const float*)d_in[2];   // (1024,1024)
    float*       out   = (float*)d_out;           // (2,2048,1024)

    const size_t NX = 4194304, NWQ = 3145728, NWO = 1048576;
    bf16* xb   = (bf16*)d_ws;
    bf16* wqb  = xb  + NX;
    bf16* wob  = wqb + NWQ;
    bf16* qkvb = wob + NWO;                  // (4096, 3072)
    bf16* aob  = qkvb + (size_t)4096 * 3072; // (4096, 1024)

    cvt_bf16_kernel<<<NX  / 2048, 256, 0, stream>>>(x,     xb);
    cvt_bf16_kernel<<<NWQ / 2048, 256, 0, stream>>>(w_qkv, wqb);
    cvt_bf16_kernel<<<NWO / 2048, 256, 0, stream>>>(w_out, wob);

    // qkv = x (4096x1024) * w_qkv^T (1024x3072), bf16 out
    gemm_wmma_kernel<true><<<dim3(24, 32), 256, 0, stream>>>(
        xb, wqb, (void*)qkvb, 4096, 3072, 1024);

    // attention: grid (S/128, B*H)
    attn_kernel<<<dim3(16, 32), 256, 0, stream>>>(qkvb, aob);

    // out = attn_out (4096x1024) * w_out^T (1024x1024), f32 out
    gemm_wmma_kernel<false><<<dim3(8, 32), 256, 0, stream>>>(
        aob, wob, (void*)out, 4096, 1024, 1024);
}
